// GraphAutoencoder_60258391163315
// MI455X (gfx1250) — matrix operand order repaired and checked by reading, not compiled, not run on hardware
//
#include <hip/hip_runtime.h>

// ---------------- model dims ----------------
#define T_FR   48
#define NA_N   1024
#define ND_N   1024
#define NB_N   128
#define IN_F   16
#define H_F    128
#define HT_F   256
#define OUT_F  128
#define E_AA_N 8192
#define E_AD_N 8192
#define E_DD_N 8192
#define E_AB_N 2048
#define E_DB_N 2048

typedef __attribute__((ext_vector_type(16))) _Float16 v16h;
typedef __attribute__((ext_vector_type(8)))  float    v8f;

#if __has_builtin(__builtin_amdgcn_tensor_load_to_lds)
#define HAS_TDM 1
#endif

#ifdef HAS_TDM
typedef __attribute__((ext_vector_type(4))) unsigned int u32x4;
typedef __attribute__((ext_vector_type(4))) int          i32x4;
typedef __attribute__((ext_vector_type(8))) int          i32x8;

// 1D contiguous tile: copy n_elems f32 from global -> LDS via the Tensor Data Mover.
// D# layout per CDNA5 ISA (2D groups only, VADDR2/3 zeroed).
__device__ __forceinline__ void tdm_load_1d(unsigned int lds_addr, const void* gptr,
                                            unsigned int n_elems) {
  unsigned long long ga = (unsigned long long)(__SIZE_TYPE__)gptr;
  u32x4 g0;
  g0[0] = 1u;                                   // count=1, user mode, no gather
  g0[1] = lds_addr;                             // LDS byte address
  g0[2] = (unsigned int)ga;                     // global_addr[31:0]
  g0[3] = (unsigned int)((ga >> 32) & 0x1FFFFFFu) | (2u << 30);  // addr[56:32] | type=2
  i32x8 g1;
  g1[0] = (int)(2u << 16);                      // data_size=2 -> 4-byte elements
  g1[1] = (int)((n_elems & 0xFFFFu) << 16);     // tensor_dim0[15:0]
  g1[2] = (int)(((n_elems >> 16) & 0xFFFFu) | (1u << 16));  // tensor_dim0 hi | tensor_dim1=1
  g1[3] = (int)((n_elems & 0xFFFFu) << 16);     // tile_dim0 = n_elems
  g1[4] = 0;                                    // tile_dim1 = tile_dim2 = 0 (1D tile)
  g1[5] = (int)n_elems;                         // tensor_dim0_stride lo
  g1[6] = 0;
  g1[7] = 0;
  i32x4 zz = {0, 0, 0, 0};
#if __clang_major__ >= 23
  i32x8 z8 = {0, 0, 0, 0, 0, 0, 0, 0};
  __builtin_amdgcn_tensor_load_to_lds(g0, g1, zz, zz, z8, 0);
#else
  __builtin_amdgcn_tensor_load_to_lds(g0, g1, zz, zz, 0);
#endif
}
#endif  // HAS_TDM

// ---------------- WMMA fragment helpers (16x16x32 f16, f32 acc) ----------------
// A 16x32 (ISA 7.12.2): lane -> row m=lane&15, half=lane>>4; VGPR p holds
// K = (p>=4?16:0) + half*8 + (p&3)*2 + {0,1}
__device__ __forceinline__ v16h frag_a_from(const float* rowptr, int half) {
  v16h a;
#pragma unroll
  for (int p = 0; p < 8; ++p) {
    int kb = ((p >= 4) ? 16 : 0) + half * 8 + (p & 3) * 2;
    float2 f2 = *(const float2*)(rowptr + kb);
    a[2 * p]     = (_Float16)f2.x;
    a[2 * p + 1] = (_Float16)f2.y;
  }
  return a;
}

// A fragment from an f16 row (LDS hidden-state mirror): no converts.
__device__ __forceinline__ v16h frag_a_h16(const _Float16* rowptr, int half) {
  v16h a;
#pragma unroll
  for (int p = 0; p < 8; ++p) {
    int kb = ((p >= 4) ? 16 : 0) + half * 8 + (p & 3) * 2;
    a[2 * p]     = rowptr[kb];
    a[2 * p + 1] = rowptr[kb + 1];
  }
  return a;
}

// B fragment from pre-swizzled f16 weights: one contiguous 32B load per lane.
__device__ __forceinline__ v16h frag_b16(const _Float16* __restrict__ Wswz, int frag,
                                         int lane) {
  return *(const v16h*)(Wswz + (((size_t)frag * 32 + lane) << 4));
}

__device__ __forceinline__ v8f wmma_f32(v16h a, v16h b, v8f c) {
  return __builtin_amdgcn_wmma_f32_16x16x32_f16(false, a, false, b, (short)0, c,
                                                false, false);
}

__device__ __forceinline__ float sigmf(float x) { return 1.0f / (1.0f + __expf(-x)); }

// ---------------- weight pre-swizzle: W[K,N] f32 -> fragment-ready f16 ----------------
// out[((kt*(N/16)+nt)*32 + lane)*16 + e] = W[kt*32 + (e>>3)*16 + (lane>>4)*8 + (e&7)][nt*16 + (lane&15)]
__global__ void swizzle_w_kernel(const float* __restrict__ W, _Float16* __restrict__ out,
                                 int K, int N) {
  int idx = blockIdx.x * 256 + threadIdx.x;
  if (idx >= K * N) return;
  int e = idx & 15;
  int lane = (idx >> 4) & 31;
  int frag = idx >> 9;
  int nt16 = N >> 4;
  int nt = frag % nt16, kt = frag / nt16;
  int k = kt * 32 + (e >> 3) * 16 + (lane >> 4) * 8 + (e & 7);
  int col = nt * 16 + (lane & 15);
  out[idx] = (_Float16)W[(size_t)k * N + col];
}

// ---------------- elementwise / scatter kernels ----------------
__global__ void zero_kernel(float* __restrict__ buf, size_t n) {
  size_t i = (size_t)blockIdx.x * blockDim.x + threadIdx.x;
  if (i < n) buf[i] = 0.0f;
}

__global__ void fill_bias_kernel(float* __restrict__ buf, const float* __restrict__ b,
                                 size_t total) {
  size_t i = (size_t)blockIdx.x * blockDim.x + threadIdx.x;
  if (i < total) buf[i] = b[i & (H_F - 1)];
}

__global__ void proj_kernel(const float* __restrict__ x, const float* __restrict__ W,
                            const float* __restrict__ b, float* __restrict__ out,
                            size_t rows) {
  size_t idx = (size_t)blockIdx.x * blockDim.x + threadIdx.x;
  if (idx >= rows * H_F) return;
  size_t r = idx >> 7;
  int h = (int)(idx & (H_F - 1));
  const float* xr = x + r * IN_F;
  float acc = b[h];
#pragma unroll
  for (int k = 0; k < IN_F; ++k) acc += xr[k] * W[k * H_F + h];
  out[idx] = acc;
}

__global__ void count_kernel(const int* __restrict__ ei, int E, float* __restrict__ cnt) {
  int e = blockIdx.x * blockDim.x + threadIdx.x;
  if (e < E) atomicAdd(&cnt[ei[E + e]], 1.0f);
}

__global__ void gcnnorm_kernel(const int* __restrict__ ei, const float* __restrict__ deg,
                               float* __restrict__ nrm, int E) {
  int e = blockIdx.x * blockDim.x + threadIdx.x;
  if (e >= E) return;
  int s = ei[e], d = ei[E + e];
  float ds_ = deg[s] > 0.0f ? rsqrtf(fmaxf(deg[s], 1.0f)) : 0.0f;
  float dd_ = deg[d] > 0.0f ? rsqrtf(fmaxf(deg[d], 1.0f)) : 0.0f;
  nrm[e] = ds_ * dd_;
}

// z[t, dst, h] += src[t, srcnode, h] * (nrm ? nrm[e] : 1); grid = (E, T), block = H
__global__ void scatter_kernel(const float* __restrict__ srcbuf, const int* __restrict__ ei,
                               const float* __restrict__ nrm, float* __restrict__ dstbuf,
                               int E, int Ns, int Nd) {
  int e = blockIdx.x, t = blockIdx.y, h = threadIdx.x;
  int s = ei[e], d = ei[E + e];
  float w = nrm ? nrm[e] : 1.0f;
  float v = srcbuf[((size_t)t * Ns + s) * H_F + h] * w;
  atomicAdd(&dstbuf[((size_t)t * Nd + d) * H_F + h], v);
}

__global__ void meandiv_kernel(float* __restrict__ buf, const float* __restrict__ cnt,
                               int Nd, size_t total) {
  size_t i = (size_t)blockIdx.x * blockDim.x + threadIdx.x;
  if (i >= total) return;
  int n = (int)((i >> 7) % (size_t)Nd);
  buf[i] /= fmaxf(cnt[n], 1.0f);
}

// ---------------- WMMA GEMM (N=K=128): C = (acc? C : 0) + A@W (+bias) ----------------
// grid = (M/16, 2), block = 128 (4 waves), wave -> one 16x16 tile.
__global__ void gemm_wmma_kernel(const float* __restrict__ A,
                                 const _Float16* __restrict__ Wswz,
                                 const float* __restrict__ bias, float* __restrict__ C,
                                 int M, int accumulate) {
  const int N = H_F, NT = H_F / 16;
  int lane = threadIdx.x & 31;
  int wave = threadIdx.x >> 5;
  int m0 = blockIdx.x * 16;
  int nt = blockIdx.y * 4 + wave;
  int n0 = nt * 16;
  int half = lane >> 4;
  int mr = lane & 15;
  v8f acc = {};
  if (accumulate) {
#pragma unroll
    for (int i = 0; i < 8; ++i)
      acc[i] = C[(size_t)(m0 + half * 8 + i) * N + n0 + mr];
  }
  const float* arow = A + (size_t)(m0 + mr) * H_F;
  __builtin_prefetch(arow + 16 * H_F, 0, 1);  // next M-tile's row -> global_prefetch_b8
#pragma unroll
  for (int kt = 0; kt < 4; ++kt) {
    v16h a = frag_a_from(arow + kt * 32, half);
    v16h b = frag_b16(Wswz, kt * NT + nt, lane);
    acc = wmma_f32(a, b, acc);
  }
  float bv = bias ? bias[n0 + mr] : 0.0f;
#pragma unroll
  for (int i = 0; i < 8; ++i)
    C[(size_t)(m0 + half * 8 + i) * N + n0 + mr] = acc[i] + bv;
}

// ---------------- fused GRU: 48 steps; h in LDS (f32 + f16 mirror); z via TDM ----------
// block = 256 (8 waves) owns 16 node rows; wave w owns hidden dims [w*32, w*32+32):
// its r/z/n gate columns (j, 256+j, 512+j) live entirely in its own WMMA tiles.
__global__ void gru_kernel(const float* __restrict__ z, const _Float16* __restrict__ Wi16,
                           const _Float16* __restrict__ Wh16, const float* __restrict__ bi,
                           const float* __restrict__ bh, float* __restrict__ hT,
                           int Nn, int Tn) {
  __shared__ float    hbuf[16][HT_F];        // 16 KB
  __shared__ _Float16 hbuf16[16][HT_F];      // 8 KB f16 mirror for A fragments
  __shared__ float    zstage[2][16][H_F];    // 16 KB double-buffered z tile
  const int NT = 48;                         // 768/16 column tiles
  int tid = threadIdx.x;
  int lane = tid & 31, wave = tid >> 5;
  int half = lane >> 4, mr = lane & 15;
  int m0 = blockIdx.x * 16;

  for (int i = tid; i < 16 * HT_F; i += 256) {
    (&hbuf[0][0])[i] = 0.0f;
    (&hbuf16[0][0])[i] = (_Float16)0.0f;
  }
#ifdef HAS_TDM
  if (wave == 0) {
    tdm_load_1d((unsigned int)(__SIZE_TYPE__)&zstage[0][0][0],
                z + (size_t)m0 * H_F, 16 * H_F);
    __builtin_amdgcn_s_wait_tensorcnt(0);
  }
#endif
  __syncthreads();

  for (int t = 0; t < Tn; ++t) {
#ifdef HAS_TDM
    int buf = t & 1;
    if (wave == 0 && (t + 1) < Tn)
      tdm_load_1d((unsigned int)(__SIZE_TYPE__)&zstage[(t + 1) & 1][0][0],
                  z + ((size_t)(t + 1) * Nn + m0) * H_F, 16 * H_F);
#else
    int buf = 0;
    const float* zsrc = z + ((size_t)t * Nn + m0) * H_F;
    for (int i = tid; i < 16 * H_F; i += 256) (&zstage[0][0][0])[i] = zsrc[i];
    __syncthreads();
#endif
    const float* ztile = &zstage[buf][0][0];
    float hnew[2][8];
#pragma unroll 1
    for (int c = 0; c < 2; ++c) {
      int ct = wave * 2 + c;            // column tile within each gate (0..15)
      int jr = ct * 16, jz = 256 + jr, jn = 512 + jr;
      v8f gr = {}, gz = {}, gn = {}, hr = {}, hz = {}, hn = {};
#pragma unroll
      for (int kt = 0; kt < 4; ++kt) {  // gi = z_t @ Wi (K=128)
        v16h a = frag_a_from(ztile + mr * H_F + kt * 32, half);
        gr = wmma_f32(a, frag_b16(Wi16, kt * NT + ct, lane), gr);
        gz = wmma_f32(a, frag_b16(Wi16, kt * NT + 16 + ct, lane), gz);
        gn = wmma_f32(a, frag_b16(Wi16, kt * NT + 32 + ct, lane), gn);
      }
#pragma unroll
      for (int kt = 0; kt < 8; ++kt) {  // gh = h @ Wh (K=256), A from f16 LDS mirror
        v16h a = frag_a_h16(&hbuf16[mr][kt * 32], half);
        hr = wmma_f32(a, frag_b16(Wh16, kt * NT + ct, lane), hr);
        hz = wmma_f32(a, frag_b16(Wh16, kt * NT + 16 + ct, lane), hz);
        hn = wmma_f32(a, frag_b16(Wh16, kt * NT + 32 + ct, lane), hn);
      }
      float bir = bi[jr + mr], biz = bi[jz + mr], bin = bi[jn + mr];
      float bhr = bh[jr + mr], bhz = bh[jz + mr], bhn = bh[jn + mr];
#pragma unroll
      for (int i = 0; i < 8; ++i) {
        int row = half * 8 + i;
        float r  = sigmf(gr[i] + bir + hr[i] + bhr);
        float zg = sigmf(gz[i] + biz + hz[i] + bhz);
        float nc = tanhf(gn[i] + bin + r * (hn[i] + bhn));
        float hold = hbuf[row][jr + mr];
        hnew[c][i] = (1.0f - zg) * nc + zg * hold;
      }
    }
    __syncthreads();  // all reads of h done
#pragma unroll
    for (int c = 0; c < 2; ++c)
#pragma unroll
      for (int i = 0; i < 8; ++i) {
        int row = half * 8 + i, j = (wave * 2 + c) * 16 + mr;
        hbuf[row][j]   = hnew[c][i];
        hbuf16[row][j] = (_Float16)hnew[c][i];
      }
#ifdef HAS_TDM
    if (wave == 0) __builtin_amdgcn_s_wait_tensorcnt(0);  // t+1 tile landed
#endif
    __syncthreads();
  }
  for (int i = tid; i < 16 * HT_F; i += 256) {
    int rr = i / HT_F, cc = i % HT_F;
    hT[(size_t)(m0 + rr) * HT_F + cc] = hbuf[rr][cc];
  }
}

// ---------------- pooling + output head ----------------
__global__ void pool_kernel(const float* __restrict__ hA, const float* __restrict__ hD,
                            const float* __restrict__ hB, float* __restrict__ pooled) {
  int b = blockIdx.x;
  int type = b >> 8, j = b & 255;
  const float* src;
  int n;
  if (type == 0)      { src = hA; n = NA_N; }
  else if (type == 1) { src = hD; n = ND_N; }
  else                { src = hB; n = NB_N; }
  __shared__ float red[256];
  float s = 0.0f;
  for (int i = threadIdx.x; i < n; i += 256) s += src[(size_t)i * HT_F + j];
  red[threadIdx.x] = s;
  __syncthreads();
  for (int st = 128; st > 0; st >>= 1) {
    if (threadIdx.x < st) red[threadIdx.x] += red[threadIdx.x + st];
    __syncthreads();
  }
  if (threadIdx.x == 0) pooled[b] = red[0] / (float)n;
}

__global__ void out_kernel(const float* __restrict__ pooled, const float* __restrict__ Wout,
                           const float* __restrict__ bout, float* __restrict__ out) {
  int o = threadIdx.x;
  float acc = bout[o];
  for (int j = 0; j < 3 * HT_F; ++j) acc += pooled[j] * Wout[j * OUT_F + o];
  out[o] = acc;
}

// ---------------- host-side orchestration ----------------
static inline void launch_gemm(const float* A, const _Float16* Wswz, const float* bias,
                               float* C, int M, int acc, hipStream_t s) {
  dim3 g(M / 16, 2), b(128);
  gemm_wmma_kernel<<<g, b, 0, s>>>(A, Wswz, bias, C, M, acc);
}

extern "C" void kernel_launch(void* const* d_in, const int* in_sizes, int n_in,
                              void* d_out, int out_size, void* d_ws, size_t ws_size,
                              hipStream_t stream) {
  const float* x_attk = (const float*)d_in[0];
  const float* x_def  = (const float*)d_in[1];
  const float* x_ball = (const float*)d_in[2];
  const int* ei_aa = (const int*)d_in[3];
  const int* ei_ad = (const int*)d_in[4];
  const int* ei_dd = (const int*)d_in[5];
  const int* ei_ab = (const int*)d_in[6];
  const int* ei_db = (const int*)d_in[7];
  const float* Wp_a = (const float*)d_in[8];  const float* bp_a = (const float*)d_in[9];
  const float* Wp_d = (const float*)d_in[10]; const float* bp_d = (const float*)d_in[11];
  const float* Wp_b = (const float*)d_in[12]; const float* bp_b = (const float*)d_in[13];
  const float* Wg_aa = (const float*)d_in[14]; const float* bg_aa = (const float*)d_in[15];
  const float* Wg_dd = (const float*)d_in[16]; const float* bg_dd = (const float*)d_in[17];
  const float* Wl_ad = (const float*)d_in[18]; const float* Wr_ad = (const float*)d_in[19];
  const float* bl_ad = (const float*)d_in[20];
  const float* Wl_ab = (const float*)d_in[21]; const float* Wr_ab = (const float*)d_in[22];
  const float* bl_ab = (const float*)d_in[23];
  const float* Wl_db = (const float*)d_in[24]; const float* Wr_db = (const float*)d_in[25];
  const float* bl_db = (const float*)d_in[26];
  const float* Wi = (const float*)d_in[27]; const float* Wh = (const float*)d_in[28];
  const float* bi = (const float*)d_in[29]; const float* bh = (const float*)d_in[30];
  const float* Wout = (const float*)d_in[31]; const float* bout = (const float*)d_in[32];

  // ---------- workspace layout (f32 region, then 32B-aligned f16 region) ----------
  const size_t SZ_P  = (size_t)T_FR * NA_N * H_F;
  const size_t SZ_PB = (size_t)T_FR * NB_N * H_F;
  float* ws = (float*)d_ws;
  float* pa  = ws;
  float* pd  = pa + SZ_P;
  float* pb  = pd + SZ_P;
  float* za  = pb + SZ_PB;
  float* zd  = za + SZ_P;
  float* zb  = zd + SZ_P;
  float* tmp = zb + SZ_PB;
  float* deg = tmp + SZ_P;
  float* nrm = deg + 1024;
  float* hta = nrm + 8192;
  float* htd = hta + (size_t)NA_N * HT_F;
  float* htb = htd + (size_t)ND_N * HT_F;
  float* pooled = htb + (size_t)NB_N * HT_F;
  size_t h16off = (size_t)(pooled + 768 - ws);
  h16off = (h16off + 7) & ~(size_t)7;          // 32B alignment for v16h loads
  _Float16* h16 = (_Float16*)(ws + h16off);
  const size_t SW_HH = (size_t)H_F * H_F;      // 16384 halves per 128x128 weight
  _Float16* wg_aa16 = h16;
  _Float16* wg_dd16 = wg_aa16 + SW_HH;
  _Float16* wl_ad16 = wg_dd16 + SW_HH;
  _Float16* wr_ad16 = wl_ad16 + SW_HH;
  _Float16* wl_ab16 = wr_ad16 + SW_HH;
  _Float16* wr_ab16 = wl_ab16 + SW_HH;
  _Float16* wl_db16 = wr_ab16 + SW_HH;
  _Float16* wr_db16 = wl_db16 + SW_HH;
  _Float16* wi16    = wr_db16 + SW_HH;                       // 128x768
  _Float16* wh16    = wi16 + (size_t)H_F * 3 * HT_F;         // 256x768

  const int TRA = T_FR * NA_N, TRD = T_FR * ND_N, TRB = T_FR * NB_N;
  auto blks = [](size_t n) { return (unsigned)((n + 255) / 256); };

  // 0) pre-swizzle all GEMM weights to fragment-ready f16
  swizzle_w_kernel<<<blks(SW_HH), 256, 0, stream>>>(Wg_aa, wg_aa16, H_F, H_F);
  swizzle_w_kernel<<<blks(SW_HH), 256, 0, stream>>>(Wg_dd, wg_dd16, H_F, H_F);
  swizzle_w_kernel<<<blks(SW_HH), 256, 0, stream>>>(Wl_ad, wl_ad16, H_F, H_F);
  swizzle_w_kernel<<<blks(SW_HH), 256, 0, stream>>>(Wr_ad, wr_ad16, H_F, H_F);
  swizzle_w_kernel<<<blks(SW_HH), 256, 0, stream>>>(Wl_ab, wl_ab16, H_F, H_F);
  swizzle_w_kernel<<<blks(SW_HH), 256, 0, stream>>>(Wr_ab, wr_ab16, H_F, H_F);
  swizzle_w_kernel<<<blks(SW_HH), 256, 0, stream>>>(Wl_db, wl_db16, H_F, H_F);
  swizzle_w_kernel<<<blks(SW_HH), 256, 0, stream>>>(Wr_db, wr_db16, H_F, H_F);
  swizzle_w_kernel<<<blks((size_t)H_F * 3 * HT_F), 256, 0, stream>>>(Wi, wi16, H_F, 3 * HT_F);
  swizzle_w_kernel<<<blks((size_t)HT_F * 3 * HT_F), 256, 0, stream>>>(Wh, wh16, HT_F, 3 * HT_F);

  // 1) projections
  proj_kernel<<<blks((size_t)TRA * H_F), 256, 0, stream>>>(x_attk, Wp_a, bp_a, pa, TRA);
  proj_kernel<<<blks((size_t)TRD * H_F), 256, 0, stream>>>(x_def,  Wp_d, bp_d, pd, TRD);
  proj_kernel<<<blks((size_t)TRB * H_F), 256, 0, stream>>>(x_ball, Wp_b, bp_b, pb, TRB);

  // 2) z_a = GCN(pa, ei_aa)
  fill_bias_kernel<<<blks(SZ_P), 256, 0, stream>>>(za, bg_aa, SZ_P);
  zero_kernel<<<blks(NA_N), 256, 0, stream>>>(deg, NA_N);
  count_kernel<<<blks(E_AA_N), 256, 0, stream>>>(ei_aa, E_AA_N, deg);
  gcnnorm_kernel<<<blks(E_AA_N), 256, 0, stream>>>(ei_aa, deg, nrm, E_AA_N);
  launch_gemm(pa, wg_aa16, nullptr, tmp, TRA, 0, stream);
  scatter_kernel<<<dim3(E_AA_N, T_FR), H_F, 0, stream>>>(tmp, ei_aa, nrm, za, E_AA_N, NA_N, NA_N);

  // 3) z_d = GCN(pd, ei_dd) + SAGE(pa->pd, ei_ad)
  fill_bias_kernel<<<blks(SZ_P), 256, 0, stream>>>(zd, bg_dd, SZ_P);
  zero_kernel<<<blks(ND_N), 256, 0, stream>>>(deg, ND_N);
  count_kernel<<<blks(E_DD_N), 256, 0, stream>>>(ei_dd, E_DD_N, deg);
  gcnnorm_kernel<<<blks(E_DD_N), 256, 0, stream>>>(ei_dd, deg, nrm, E_DD_N);
  launch_gemm(pd, wg_dd16, nullptr, tmp, TRD, 0, stream);
  scatter_kernel<<<dim3(E_DD_N, T_FR), H_F, 0, stream>>>(tmp, ei_dd, nrm, zd, E_DD_N, ND_N, ND_N);

  zero_kernel<<<blks(ND_N), 256, 0, stream>>>(deg, ND_N);
  count_kernel<<<blks(E_AD_N), 256, 0, stream>>>(ei_ad, E_AD_N, deg);
  zero_kernel<<<blks(SZ_P), 256, 0, stream>>>(tmp, SZ_P);
  scatter_kernel<<<dim3(E_AD_N, T_FR), H_F, 0, stream>>>(pa, ei_ad, nullptr, tmp, E_AD_N, NA_N, ND_N);
  meandiv_kernel<<<blks(SZ_P), 256, 0, stream>>>(tmp, deg, ND_N, SZ_P);
  launch_gemm(tmp, wl_ad16, bl_ad, zd, TRD, 1, stream);
  launch_gemm(pd,  wr_ad16, nullptr, zd, TRD, 1, stream);

  // 4) z_b = SAGE(pa->pb, ei_ab) + SAGE(pd->pb, ei_db)
  zero_kernel<<<blks(NB_N), 256, 0, stream>>>(deg, NB_N);
  count_kernel<<<blks(E_AB_N), 256, 0, stream>>>(ei_ab, E_AB_N, deg);
  zero_kernel<<<blks(SZ_PB), 256, 0, stream>>>(tmp, SZ_PB);
  scatter_kernel<<<dim3(E_AB_N, T_FR), H_F, 0, stream>>>(pa, ei_ab, nullptr, tmp, E_AB_N, NA_N, NB_N);
  meandiv_kernel<<<blks(SZ_PB), 256, 0, stream>>>(tmp, deg, NB_N, SZ_PB);
  launch_gemm(tmp, wl_ab16, bl_ab, zb, TRB, 0, stream);
  launch_gemm(pb,  wr_ab16, nullptr, zb, TRB, 1, stream);

  zero_kernel<<<blks(NB_N), 256, 0, stream>>>(deg, NB_N);
  count_kernel<<<blks(E_DB_N), 256, 0, stream>>>(ei_db, E_DB_N, deg);
  zero_kernel<<<blks(SZ_PB), 256, 0, stream>>>(tmp, SZ_PB);
  scatter_kernel<<<dim3(E_DB_N, T_FR), H_F, 0, stream>>>(pd, ei_db, nullptr, tmp, E_DB_N, ND_N, NB_N);
  meandiv_kernel<<<blks(SZ_PB), 256, 0, stream>>>(tmp, deg, NB_N, SZ_PB);
  launch_gemm(tmp, wl_db16, bl_db, zb, TRB, 1, stream);
  launch_gemm(pb,  wr_db16, nullptr, zb, TRB, 1, stream);

  // 5) fused GRUs (h in LDS for all 48 steps; z tiles DMA'd in via TDM)
  gru_kernel<<<NA_N / 16, 256, 0, stream>>>(za, wi16, wh16, bi, bh, hta, NA_N, T_FR);
  gru_kernel<<<ND_N / 16, 256, 0, stream>>>(zd, wi16, wh16, bi, bh, htd, ND_N, T_FR);
  gru_kernel<<<NB_N / 16, 256, 0, stream>>>(zb, wi16, wh16, bi, bh, htb, NB_N, T_FR);

  // 6) pool over nodes + output head
  pool_kernel<<<3 * HT_F, 256, 0, stream>>>(hta, htd, htb, pooled);
  out_kernel<<<1, OUT_F, 0, stream>>>(pooled, Wout, bout, (float*)d_out);
}
